// SNNTorchLeaky_23450521436446
// MI455X (gfx1250) — matrix operand order repaired and checked by reading, not compiled, not run on hardware
//
#include <hip/hip_runtime.h>

// ---- types --------------------------------------------------------------
typedef __attribute__((ext_vector_type(16))) __bf16        v16bf;
typedef __attribute__((ext_vector_type(8)))  float         v8f;
typedef __attribute__((ext_vector_type(4)))  float         f32x4;
typedef __attribute__((ext_vector_type(4)))  unsigned int  u32x4;
typedef __attribute__((ext_vector_type(2)))  unsigned int  u32x2;
typedef unsigned short u16;

union FragU { v16bf v; u32x4 q[2]; };

#define B_   512
#define T_   100
#define NI_  2048
#define NH_  4096
#define NO_  128
#define BETA_ 0.9048374180359595f   // exp(-0.1)

__device__ __forceinline__ u16 f2bf(float f) {
    union { float f; unsigned u; } v; v.f = f;
    unsigned r = v.u + 0x7FFFu + ((v.u >> 16) & 1u);   // RNE
    return (u16)(r >> 16);
}

// ---- kernel 1: f32 -> bf16 conversion (vectorized x4) -------------------
__global__ void cvt_f32_bf16(const float* __restrict__ src,
                             u16* __restrict__ dst, int n4) {
    int i = blockIdx.x * blockDim.x + threadIdx.x;
    if (i >= n4) return;
    f32x4 f = ((const f32x4*)src)[i];
    unsigned p0 = (unsigned)f2bf(f[0]) | ((unsigned)f2bf(f[1]) << 16);
    unsigned p1 = (unsigned)f2bf(f[2]) | ((unsigned)f2bf(f[3]) << 16);
    ((u32x2*)dst)[i] = (u32x2){p0, p1};
}

// ---- kernel 2: zero init ------------------------------------------------
__global__ void zero_f32(float* __restrict__ p, int n4) {
    int i = blockIdx.x * blockDim.x + threadIdx.x;
    if (i >= n4) return;
    ((f32x4*)p)[i] = (f32x4){0.f, 0.f, 0.f, 0.f};
}

// ---- kernel 3: cur1_all = einsum('bti,hi->tbh', x, W1) + b1 -------------
// M = T*B = 51200 (row r = t*B + b), N = NH = 4096, K = NI = 2048.
// WG: 512 threads = 16 waves; 128x128 tile; wave grid 8(M) x 2(N);
// each wave 16x64 = 1 A-frag x 4 B-frags = 4 WMMA per 32-K step (4 v8f acc).
// Double-buffered LDS (loop-carried XOR offset, unroll disabled so the body
// is iteration-invariant and accumulators stay pinned), one barrier/K-step.
#define LDA_ 40              // 32 K + 8 pad (u16): 80B rows, 16B aligned
#define BUF_ (128 * LDA_)    // one LDS buffer (u16 elements)
#define NK_  (NI_ / 32)      // 64 K-steps
__global__ __launch_bounds__(512)
void gemm1(const u16* __restrict__ xb, const u16* __restrict__ w1b,
           const float* __restrict__ b1, float* __restrict__ cur1) {
    __shared__ u16 As[2 * BUF_];
    __shared__ u16 Bs[2 * BUF_];

    const int tid  = threadIdx.x;
    const int lane = tid & 31;
    const int w    = tid >> 5;        // 0..15
    const int wm   = w & 7;           // M-slice of wave (0..7), 16 rows each
    const int wn   = w >> 3;          // N-slice of wave (0..1), 64 cols each
    const int mbase = blockIdx.x * 128;
    const int nbase = blockIdx.y * 128;

    // staging map: 512 16B-segments per tile, one per thread
    const int r0 = tid >> 2, q0 = (tid & 3) * 8;       // rows 0..127
    const int ga0 = mbase + r0;                        // A row (r = t*B + b)
    const u16* asrc = xb + ((size_t)(ga0 % B_) * T_ + (ga0 / B_)) * NI_ + q0;
    const u16* bsrc = w1b + (size_t)(nbase + r0) * NI_ + q0;
    const int d0 = r0 * LDA_ + q0;

    // prologue: stage K-step 0 into buffer 0
    {
        u32x4 a0 = *(const u32x4*)asrc;
        u32x4 c0 = *(const u32x4*)bsrc;
        *(u32x4*)(As + d0) = a0;
        *(u32x4*)(Bs + d0) = c0;
    }
    __syncthreads();

    v8f acc[4] = {v8f{}, v8f{}, v8f{}, v8f{}};
    const int koffA = (lane < 16) ? 0 : 8;    // A frag K-offset
    const int koffB = (lane < 16) ? 0 : 8;    // B frag K-offset
    const int aoff  = (wm * 16 + (lane & 15)) * LDA_ + koffA;
    const int boff  = (wn * 64 + (lane & 15)) * LDA_ + koffB;

    unsigned co = 0;   // current LDS buffer offset (u16 elements), toggled
#pragma unroll 1
    for (int kt = 0; kt < NK_ - 1; ++kt) {
        // issue next tile's global loads early (latency hidden by compute)
        const int k = (kt + 1) * 32;
        u32x4 a0 = *(const u32x4*)(asrc + k);
        u32x4 c0 = *(const u32x4*)(bsrc + k);

        // compute from current buffer
        const u16* Ac = As + co;
        const u16* Bc = Bs + co;
        FragU fa;
        fa.q[0] = *(const u32x4*)(Ac + aoff);        // K {0..7} | {8..15}
        fa.q[1] = *(const u32x4*)(Ac + aoff + 16);   // K {16..23} | {24..31}
#pragma unroll
        for (int bs = 0; bs < 4; ++bs) {
            FragU fb;
            fb.q[0] = *(const u32x4*)(Bc + boff + bs * (16 * LDA_));
            fb.q[1] = *(const u32x4*)(Bc + boff + bs * (16 * LDA_) + 16);
            acc[bs] = __builtin_amdgcn_wmma_f32_16x16x32_bf16(
                false, fa.v, false, fb.v, (short)0, acc[bs], false, false);
        }

        // store next tile into the other buffer; single barrier per K-step
        co ^= BUF_;
        *(u32x4*)(As + co + d0) = a0;
        *(u32x4*)(Bs + co + d0) = c0;
        __syncthreads();
    }
    // final K-step: compute only
    {
        const u16* Ac = As + co;
        const u16* Bc = Bs + co;
        FragU fa;
        fa.q[0] = *(const u32x4*)(Ac + aoff);
        fa.q[1] = *(const u32x4*)(Ac + aoff + 16);
#pragma unroll
        for (int bs = 0; bs < 4; ++bs) {
            FragU fb;
            fb.q[0] = *(const u32x4*)(Bc + boff + bs * (16 * LDA_));
            fb.q[1] = *(const u32x4*)(Bc + boff + bs * (16 * LDA_) + 16);
            acc[bs] = __builtin_amdgcn_wmma_f32_16x16x32_bf16(
                false, fa.v, false, fb.v, (short)0, acc[bs], false, false);
        }
    }

    // epilogue: C layout — VGPR j: M = j (+8 for lanes>=16), N = lane%16
    const int rb = mbase + wm * 16 + ((lane < 16) ? 0 : 8);
#pragma unroll
    for (int bs = 0; bs < 4; ++bs) {
        const int col = nbase + wn * 64 + bs * 16 + (lane & 15);
        const float bias = b1[col];
#pragma unroll
        for (int j = 0; j < 8; ++j)
            cur1[(size_t)(rb + j) * NH_ + col] = acc[bs][j] + bias;
    }
}

// ---- kernel 4: one SNN time-step (mem1 update + spk1 + layer2 WMMA + mem2)
// 32 WGs x 256 thr; WG owns 16 batch rows. NH processed in 512-wide chunks:
// elementwise update -> spikes (bf16) to LDS -> WMMA accumulate 16x128.
#define CH_  512
#define LDS_ (CH_ + 8)   // u16 pitch: 1040B rows (16B aligned)
__global__ __launch_bounds__(256)
void snn_step(const float* __restrict__ cur1, const u16* __restrict__ w2b,
              const float* __restrict__ b2, float* __restrict__ mem1,
              float* __restrict__ mem2, float* __restrict__ out, int t) {
    __shared__ u16 spk[16 * LDS_];

    const int tid  = threadIdx.x;
    const int lane = tid & 31;
    const int w    = tid >> 5;
    const int bbase = blockIdx.x * 16;

    v8f acc = {};
    const int koffB = (lane < 16) ? 0 : 8;
    const int koffA = (lane < 16) ? 0 : 8;
    const u16* apLds = spk + (lane & 15) * LDS_ + koffA;
    const u16* w2row = w2b + (size_t)(w * 16 + (lane & 15)) * NH_ + koffB;

    for (int c = 0; c < NH_ / CH_; ++c) {
        // phase A: leaky-integrate 16 x 512 slice, spikes -> LDS
#pragma unroll
        for (int it = 0; it < 8; ++it) {
            int q    = tid + it * 256;          // float4 index in [0,2048)
            int row  = q >> 7;                  // 128 float4 per row
            int colq = q & 127;
            int h    = c * CH_ + colq * 4;
            size_t gb = (size_t)(bbase + row);
            f32x4 cu = *(const f32x4*)(cur1 + ((size_t)t * B_ + gb) * NH_ + h);
            f32x4 m  = *(const f32x4*)(mem1 + gb * NH_ + h);
            f32x4 nm; u16 s[4];
#pragma unroll
            for (int e = 0; e < 4; ++e) {
                float reset = (m[e] > 1.0f) ? 1.0f : 0.0f;   // uses PREVIOUS mem
                float mv = BETA_ * m[e] + cu[e] - reset;
                nm[e] = mv;
                s[e]  = (mv > 1.0f) ? (u16)0x3F80 : (u16)0;  // bf16 1.0 / 0.0
            }
            *(f32x4*)(mem1 + gb * NH_ + h) = nm;
            unsigned p0 = (unsigned)s[0] | ((unsigned)s[1] << 16);
            unsigned p1 = (unsigned)s[2] | ((unsigned)s[3] << 16);
            *(u32x2*)(spk + row * LDS_ + colq * 4) = (u32x2){p0, p1};
        }
        __syncthreads();

        // phase B: cur2 += spk1[16 x 512] @ W2^T chunk  (16 WMMA iterations)
#pragma unroll 4
        for (int k0 = 0; k0 < CH_; k0 += 32) {
            FragU a, bf;
            a.q[0]  = *(const u32x4*)(apLds + k0);
            a.q[1]  = *(const u32x4*)(apLds + k0 + 16);
            const u16* bp = w2row + c * CH_ + k0;
            bf.q[0] = *(const u32x4*)(bp);
            bf.q[1] = *(const u32x4*)(bp + 16);
            acc = __builtin_amdgcn_wmma_f32_16x16x32_bf16(
                false, a.v, false, bf.v, (short)0, acc, false, false);
        }
        __syncthreads();   // LDS reuse across chunks
    }

    // epilogue: mem2 update + record mem2/spk2
    const int o    = w * 16 + (lane & 15);
    const float bias = b2[o];
    const int mloc = (lane < 16) ? 0 : 8;
#pragma unroll
    for (int j = 0; j < 8; ++j) {
        int b = bbase + mloc + j;
        size_t idx = (size_t)b * NO_ + o;
        float old   = mem2[idx];
        float reset = (old > 1.0f) ? 1.0f : 0.0f;
        float m2    = BETA_ * old + (acc[j] + bias) - reset;
        mem2[idx] = m2;
        size_t oidx = (size_t)t * (B_ * NO_) + idx;
        out[oidx] = m2;                                         // mem2_rec
        out[(size_t)T_ * B_ * NO_ + oidx] = (m2 > 1.0f) ? 1.0f : 0.0f; // spk2_rec
    }
}

// ---- host launcher ------------------------------------------------------
extern "C" void kernel_launch(void* const* d_in, const int* in_sizes, int n_in,
                              void* d_out, int out_size, void* d_ws, size_t ws_size,
                              hipStream_t stream) {
    const float* x  = (const float*)d_in[0];   // [B,T,A,C] = [512,100,2048]
    const float* W1 = (const float*)d_in[1];   // [4096,2048]
    const float* b1 = (const float*)d_in[2];   // [4096]
    const float* W2 = (const float*)d_in[3];   // [128,4096]
    const float* b2 = (const float*)d_in[4];   // [128]
    float* out = (float*)d_out;                // mem2_rec ++ spk2_rec

    // workspace carve (256B aligned)
    char* ws = (char*)d_ws;
    size_t off = 0;
    auto carve = [&](size_t bytes) {
        void* p = ws + off;
        off = (off + bytes + 255) & ~(size_t)255;
        return p;
    };
    const size_t NX = (size_t)B_ * T_ * NI_;       // 104,857,600
    u16*   xb    = (u16*)  carve(NX * 2);
    u16*   w1b   = (u16*)  carve((size_t)NH_ * NI_ * 2);
    u16*   w2b   = (u16*)  carve((size_t)NO_ * NH_ * 2);
    float* cur1  = (float*)carve((size_t)T_ * B_ * NH_ * 4);
    float* mem1  = (float*)carve((size_t)B_ * NH_ * 4);
    float* mem2  = (float*)carve((size_t)B_ * NO_ * 4);   // contiguous after mem1

    // 1) convert inputs to bf16
    {
        int n4 = (int)(NX / 4);                                  // 26,214,400
        cvt_f32_bf16<<<n4 / 256, 256, 0, stream>>>(x, xb, n4);
        n4 = (NH_ * NI_) / 4;                                    // 2,097,152
        cvt_f32_bf16<<<n4 / 256, 256, 0, stream>>>(W1, w1b, n4);
        n4 = (NO_ * NH_) / 4;                                    // 131,072
        cvt_f32_bf16<<<n4 / 256, 256, 0, stream>>>(W2, w2b, n4);
    }
    // 2) zero membrane state (mem1 ++ mem2, contiguous)
    {
        int n4 = (B_ * NH_ + B_ * NO_) / 4;                      // 540,672
        zero_f32<<<n4 / 256, 256, 0, stream>>>(mem1, n4);
    }
    // 3) big GEMM: cur1_all[t*B+b, h]  (128x128 tiles, 16 waves, dbl-buffered)
    {
        dim3 grid((T_ * B_) / 128, NH_ / 128);                   // 400 x 32
        gemm1<<<grid, 512, 0, stream>>>(xb, w1b, b1, cur1);
    }
    // 4) sequential scan: one launch per time step (stream-ordered)
    for (int t = 0; t < T_; ++t)
        snn_step<<<B_ / 16, 256, 0, stream>>>(cur1, w2b, b2, mem1, mem2, out, t);
}
